// TrajectoryRefiner_55173149885018
// MI455X (gfx1250) — matrix operand order repaired and checked by reading, compiled, not run on hardware
//
#include <hip/hip_runtime.h>
#include <hip/hip_bf16.h>
#include <math.h>

// Problem dims
#define B_    32
#define F_    2048
#define DE_   512
#define H_    512
#define DIN_  533
#define KPAD0 544            // DIN padded to multiple of 32 for WMMA K
#define NG_   2048           // 4*H
#define TOK   (B_ * F_)      // 65536 tokens

typedef __attribute__((ext_vector_type(8)))  float        v8f;
typedef __attribute__((ext_vector_type(16))) __bf16       v16bf;
typedef __attribute__((ext_vector_type(4)))  unsigned int u32x4;
typedef __attribute__((ext_vector_type(8)))  int          i32x8;
typedef __attribute__((ext_vector_type(4)))  int          i32x4;

union BfOp { v16bf v; u32x4 u[2]; };

__device__ __forceinline__ float sigf(float x) { return 1.0f / (1.0f + __expf(-x)); }

// ---------------------------------------------------------------------------
// CDNA5 async global->LDS (ASYNCcnt) helpers, via inline asm (VFLAT async ops)
// ---------------------------------------------------------------------------
__device__ __forceinline__ void async_g2lds_b128(unsigned lds_addr, unsigned long long gaddr) {
    asm volatile("global_load_async_to_lds_b128 %0, %1, off"
                 :: "v"(lds_addr), "v"(gaddr) : "memory");
}
__device__ __forceinline__ void wait_asynccnt0() {
#if __has_builtin(__builtin_amdgcn_s_wait_asynccnt)
    __builtin_amdgcn_s_wait_asynccnt(0);
#else
    asm volatile("s_wait_asynccnt 0x0" ::: "memory");
#endif
}

// ---------------------------------------------------------------------------
// CDNA5 Tensor Data Mover: 2D tile load (bf16 elements), TENSORcnt-tracked.
// D# built per CDNA5 ISA (groups 2/3 zero => <=2D tensor).
// This toolchain exposes the 6-arg builtin: (v4u g0, v8i g1, v4i, v4i, v8i, i32 cpol).
// ---------------------------------------------------------------------------
#if __has_builtin(__builtin_amdgcn_tensor_load_to_lds)
#define HAVE_TDM 1
__device__ __forceinline__ void tdm_load_2d_bf16(unsigned lds_addr, unsigned long long gaddr,
                                                 unsigned rows, unsigned cols, unsigned stride) {
    u32x4 g0;
    g0[0] = 1u;                                               // count=1, user descriptor
    g0[1] = lds_addr;                                         // lds_addr [63:32]
    g0[2] = (unsigned)(gaddr & 0xffffffffu);                  // global_addr lo
    g0[3] = (unsigned)((gaddr >> 32) & 0x1ffffffu) | (2u << 30); // global_addr hi | type=2
    i32x8 g1 = {};
    g1[0] = (int)(1u << 16);                                  // data_size=1 (2 bytes)
    g1[1] = (int)((cols & 0xffffu) << 16);                    // tensor_dim0 lo16 @ bits[63:48]
    g1[2] = (int)(((cols >> 16) & 0xffffu) |                  // tensor_dim0 hi16
                  ((rows & 0xffffu) << 16));                  // tensor_dim1 lo16
    g1[3] = (int)(((rows >> 16) & 0xffffu) |                  // tensor_dim1 hi16
                  ((cols & 0xffffu) << 16));                  // tile_dim0 @ [127:112]
    g1[4] = (int)(rows & 0xffffu);                            // tile_dim1 (tile_dim2=0)
    g1[5] = (int)stride;                                      // tensor_dim0_stride lo32
    g1[6] = 0;
    g1[7] = 0;
    i32x4 z4 = {};
    i32x8 z8 = {};
    __builtin_amdgcn_tensor_load_to_lds(g0, g1, z4, z4, z8, 0);
}
#else
#define HAVE_TDM 0
#endif

// ---------------------------------------------------------------------------
// 0. zero the barrier counters (workspace is poisoned by the harness)
// ---------------------------------------------------------------------------
__global__ void k_zero_u32(unsigned int* p, int n) {
    int i = blockIdx.x * blockDim.x + threadIdx.x;
    if (i < n) p[i] = 0u;
}

// ---------------------------------------------------------------------------
// 1. fp32 -> bf16 weight conversion with K padding
// ---------------------------------------------------------------------------
__global__ void k_f32_to_bf16_pad(const float* __restrict__ src, __bf16* __restrict__ dst,
                                  int rows, int K, int Kp) {
    size_t i = (size_t)blockIdx.x * blockDim.x + threadIdx.x;
    size_t n = (size_t)rows * Kp;
    if (i >= n) return;
    int r = (int)(i / Kp), c = (int)(i % Kp);
    dst[i] = (__bf16)((c < K) ? src[(size_t)r * K + c] : 0.0f);
}

// ---------------------------------------------------------------------------
// 2. assemble xc = concat(context, feet_vel*contact, root_r6d[:,1:], pred_vel) as bf16
// ---------------------------------------------------------------------------
__global__ void k_assemble_xc(const float* __restrict__ ctx, const float* __restrict__ pvel,
                              const float* __restrict__ r6d, const float* __restrict__ feet,
                              const float* __restrict__ contact, __bf16* __restrict__ xc) {
    int tok = blockIdx.x;
    int b = tok / F_, f = tok % F_;
    __bf16* row = xc + (size_t)tok * KPAD0;
    for (int c = threadIdx.x; c < KPAD0; c += blockDim.x) {
        float v = 0.0f;
        if (c < DE_) {
            v = ctx[(size_t)tok * DE_ + c];
        } else if (c < DE_ + 12) {
            int q = c - DE_;
            int leg = q / 3, d = q % 3;
            float fv = 0.0f;
            if (f > 0) {
                size_t i1 = (((size_t)b * F_ + f) * 4 + leg) * 3 + d;
                size_t i0 = (((size_t)b * F_ + f - 1) * 4 + leg) * 3 + d;
                fv = (feet[i1] - feet[i0]) * 30.0f;
            }
            v = fv * contact[((size_t)b * F_ + f) * 4 + leg];
        } else if (c < DE_ + 12 + 6) {
            int q = c - (DE_ + 12);
            v = r6d[((size_t)b * (F_ + 1) + (f + 1)) * 6 + q];
        } else if (c < DIN_) {
            int q = c - (DE_ + 12 + 6);
            v = pvel[(size_t)tok * 3 + q];
        }
        row[c] = (__bf16)v;
    }
}

// ---------------------------------------------------------------------------
// 3. bf16 NT GEMM:  C[M,N] = A[M,Kp] @ W[N,Kp]^T + b_ih[N] + b_hh[N]
//    block = 256 threads (8 waves), block tile = 128M x 128N, K step 32.
//    W tile staged to LDS via global_load_async_to_lds_b128 (ASYNCcnt).
// ---------------------------------------------------------------------------
__global__ void __launch_bounds__(256)
k_gemm_bf16_nt(const __bf16* __restrict__ A, const __bf16* __restrict__ W,
               const float* __restrict__ ba, const float* __restrict__ bb,
               float* __restrict__ C, int M, int N, int Kp) {
    __shared__ __bf16 Bt[128 * 32];        // 8 KB staged W tile
    const int wave = threadIdx.x >> 5;
    const int lane = threadIdx.x & 31;
    const int n    = lane & 15;
    const int hi   = lane >> 4;
    const int m0   = blockIdx.y * 128 + wave * 16;
    const int nblk = blockIdx.x * 128;

    v8f acc[8] = {};

    for (int k0 = 0; k0 < Kp; k0 += 32) {
        // async stage of 128 rows x 16 K (k 0..7 and 16..23) of W straight into LDS
        {
            int r = threadIdx.x >> 1, half = threadIdx.x & 1;
            const __bf16* gsrc = W + (size_t)(nblk + r) * Kp + k0 + half * 16;
            unsigned ldst = (unsigned)(uintptr_t)(Bt + r * 32 + half * 16);
            async_g2lds_b128(ldst, (unsigned long long)(uintptr_t)gsrc);
        }
        wait_asynccnt0();
        __syncthreads();

        // A operand: row m = m0 + n, K chunks {hi*8 .. hi*8+7} and {16+hi*8 ..}
        BfOp a;
        const __bf16* ap = A + (size_t)(m0 + n) * Kp + k0 + hi * 8;
        __builtin_prefetch(ap + 64, 0, 1);     // -> global_prefetch_b8
        a.u[0] = *(const u32x4*)(ap);
        a.u[1] = *(const u32x4*)(ap + 16);

#pragma unroll
        for (int j = 0; j < 8; ++j) {
            BfOp bo;
            const __bf16* bp = Bt + (j * 16 + n) * 32 + hi * 8;  // K chunks {hi*8..} and {16+hi*8..}
            bo.u[0] = *(const u32x4*)(bp);
            bo.u[1] = *(const u32x4*)(bp + 16);
            acc[j] = __builtin_amdgcn_wmma_f32_16x16x32_bf16(
                false, a.v, false, bo.v, (short)0, acc[j], false, false);
        }
        __syncthreads();
    }

    const int mb = hi * 8;
#pragma unroll
    for (int j = 0; j < 8; ++j) {
        int col = nblk + j * 16 + n;
        float bias = ba[col] + bb[col];
#pragma unroll
        for (int r = 0; r < 8; ++r) {
            C[(size_t)(m0 + mb + r) * N + col] = acc[j][r] + bias;
        }
    }
}

// ---------------------------------------------------------------------------
// 4. LSTM recurrence. 32 workgroups = 2 batch-groups (16 rows) x 16 hidden
//    slices (32 h-cols each). W_hh slice (128 rows x 512 K bf16 = 128 KB) is
//    staged once by the Tensor Data Mover and stays LDS-resident for the
//    whole scan. Per step: 16x128 output via 8 waves x 16 WMMA(bf16,K=32),
//    gates, h broadcast through L2, cluster barrier + agent-scope atomic.
// ---------------------------------------------------------------------------
#define SCAN_SMEM (128*512*2 + 16*528*2 + 16*128*4 + 16*32*4)   // 158208 B

__global__ void __launch_bounds__(256)
k_lstm_scan(const float* __restrict__ xw, const __bf16* __restrict__ Whh,
            __bf16* __restrict__ hshare,      // [2 groups][16][512] bf16
            __bf16* __restrict__ hout,        // [TOK][512] bf16
            unsigned int* __restrict__ bar) { // 2 counters (one per group)
    extern __shared__ char smem[];
    __bf16* Ws = (__bf16*)smem;                                   // [128][512]
    __bf16* hA = (__bf16*)(smem + 128 * 512 * 2);                 // [16][528] padded
    float*  gb = (float*)(smem + 128 * 512 * 2 + 16 * 528 * 2);   // [16][128]
    float*  cb = gb + 16 * 128;                                   // [16][32]

    const int wg    = blockIdx.x;
    const int group = wg >> 4;     // 0..1
    const int slice = wg & 15;     // 0..15 -> h columns [slice*32, slice*32+32)
    const int tid   = threadIdx.x;
    const int wave  = tid >> 5;
    const int lane  = tid & 31;
    const int n     = lane & 15;
    const int hi    = lane >> 4;
    unsigned int* mybar = bar + group;
    __bf16* hsh = hshare + (size_t)group * 16 * 512;

    // one-time stage of this WGP's W_hh slice (4 contiguous 32x512 row blocks,
    // one per gate): TDM 2D descriptors, tracked on TENSORcnt.
#if HAVE_TDM
    if (wave == 0) {
#pragma unroll
        for (int gate = 0; gate < 4; ++gate) {
            int grow = gate * H_ + slice * 32;                 // first global row
            unsigned long long ga = (unsigned long long)(uintptr_t)(Whh + (size_t)grow * H_);
            unsigned ldsa = (unsigned)(uintptr_t)(Ws + (size_t)gate * 32 * H_);
            tdm_load_2d_bf16(ldsa, ga, /*rows=*/32, /*cols=*/H_, /*stride=*/H_);
        }
        __builtin_amdgcn_s_wait_tensorcnt(0);
    }
#else
    for (int l = wave; l < 128; l += 8) {
        int gate = l >> 5, kk = l & 31;
        int grow = gate * H_ + slice * 32 + kk;
        const u32x4* s = (const u32x4*)(Whh + (size_t)grow * H_);
        u32x4* d = (u32x4*)(Ws + (size_t)l * H_);
        d[lane]      = s[lane];
        d[lane + 32] = s[lane + 32];
    }
#endif
    for (int i = tid; i < 16 * 528; i += 256) hA[i] = (__bf16)0.0f;
    for (int i = tid; i < 16 * 32;  i += 256) cb[i] = 0.0f;
    __syncthreads();

    for (int t = 0; t < F_; ++t) {
        // each wave: one 16x16 output tile (localN = wave*16 + n), K = 512
        v8f acc = {};
#pragma unroll 4
        for (int k0 = 0; k0 < H_; k0 += 32) {
            BfOp a, bo;
            const __bf16* ap = hA + (size_t)n * 528 + k0 + hi * 8;
            a.u[0] = *(const u32x4*)(ap);
            a.u[1] = *(const u32x4*)(ap + 16);
            const __bf16* bp = Ws + (size_t)(wave * 16 + n) * H_ + k0 + hi * 8;
            bo.u[0] = *(const u32x4*)(bp);
            bo.u[1] = *(const u32x4*)(bp + 16);
            acc = __builtin_amdgcn_wmma_f32_16x16x32_bf16(
                false, a.v, false, bo.v, (short)0, acc, false, false);
        }

        // add input-projection preactivations, stage gate values in LDS
        const int localN = wave * 16 + n;
        const int gate = localN >> 5, kk = localN & 31;
        const int gcol = gate * H_ + slice * 32 + kk;
        const int mb = hi * 8;
#pragma unroll
        for (int r = 0; r < 8; ++r) {
            int m = mb + r;
            int brow = group * 16 + m;
            float g = acc[r] + xw[((size_t)brow * F_ + t) * NG_ + gcol];
            gb[m * 128 + localN] = g;
        }
        __syncthreads();

        // elementwise LSTM cell for 16 rows x 32 columns
        for (int e = tid; e < 16 * 32; e += 256) {
            int m = e >> 5, k2 = e & 31;
            float iv = gb[m * 128 +       k2];
            float fv = gb[m * 128 + 32  + k2];
            float gv = gb[m * 128 + 64  + k2];
            float ov = gb[m * 128 + 96  + k2];
            float c = sigf(fv) * cb[m * 32 + k2] + sigf(iv) * tanhf(gv);
            cb[m * 32 + k2] = c;
            float h = sigf(ov) * tanhf(c);
            int col = slice * 32 + k2;
            __bf16 hb = (__bf16)h;
            hsh[(size_t)m * H_ + col] = hb;
            int brow = group * 16 + m;
            hout[((size_t)brow * F_ + t) * H_ + col] = hb;
        }
        __threadfence();
        __syncthreads();
        if (tid == 0) {
            __hip_atomic_fetch_add(mybar, 1u, __ATOMIC_RELEASE, __HIP_MEMORY_SCOPE_AGENT);
            while (__hip_atomic_load(mybar, __ATOMIC_ACQUIRE, __HIP_MEMORY_SCOPE_AGENT)
                   < 16u * (unsigned)(t + 1)) {
                __builtin_amdgcn_s_sleep(1);
            }
        }
        __builtin_amdgcn_s_cluster_barrier();   // CDNA5 cluster split-barrier
        __syncthreads();

        // reload the full h[16,512] (all slices) for the next step
        for (int i = tid; i < 16 * 512; i += 256) {
            int m = i >> 9, c = i & 511;
            hA[m * 528 + c] = hsh[(size_t)m * H_ + c];
        }
        __syncthreads();
    }
}

// ---------------------------------------------------------------------------
// 5. decoders: one wave32 per token, 9 dot-products of length 512 + shuffles
// ---------------------------------------------------------------------------
__global__ void __launch_bounds__(256)
k_decode(const __bf16* __restrict__ h2, const float* __restrict__ r6d_in,
         const float* __restrict__ pvel_in,
         const float* __restrict__ d0w, const float* __restrict__ d0b,
         const float* __restrict__ d1w, const float* __restrict__ d1b,
         float* __restrict__ pred_root, float* __restrict__ pred_vel) {
    int warp = blockIdx.x * (blockDim.x >> 5) + (threadIdx.x >> 5);
    if (warp >= TOK) return;
    int lane = threadIdx.x & 31;
    int b = warp / F_, f = warp % F_;
    float s[9] = {0.f, 0.f, 0.f, 0.f, 0.f, 0.f, 0.f, 0.f, 0.f};
    const __bf16* h = h2 + (size_t)warp * H_;
    for (int k = lane; k < H_; k += 32) {
        float hv = (float)h[k];
#pragma unroll
        for (int j = 0; j < 6; ++j) s[j]     += hv * d0w[j * H_ + k];
#pragma unroll
        for (int j = 0; j < 3; ++j) s[6 + j] += hv * d1w[j * H_ + k];
    }
#pragma unroll
    for (int j = 0; j < 9; ++j)
        for (int off = 16; off; off >>= 1) s[j] += __shfl_down(s[j], off, 32);
    if (lane == 0) {
        size_t rr = ((size_t)b * (F_ + 1) + f + 1) * 6;
        for (int j = 0; j < 6; ++j) pred_root[rr + j] = r6d_in[rr + j] + s[j] + d0b[j];
        for (int j = 0; j < 3; ++j)
            pred_vel[(size_t)warp * 3 + j] = pvel_in[(size_t)warp * 3 + j] + s[6 + j] + d1b[j];
        if (f == 0) {
            size_t r0 = (size_t)b * (F_ + 1) * 6;
            for (int j = 0; j < 6; ++j) pred_root[r0 + j] = r6d_in[r0 + j];
        }
    }
}

// ---------------------------------------------------------------------------
// 6. r6d -> rotation matrices for all B*(F+1) frames
// ---------------------------------------------------------------------------
__global__ void k_r6d(const float* __restrict__ pr, float* __restrict__ rot) {
    int i = blockIdx.x * blockDim.x + threadIdx.x;
    if (i >= B_ * (F_ + 1)) return;
    const float* d = pr + (size_t)i * 6;
    float a1x = d[0], a1y = d[1], a1z = d[2];
    float a2x = d[3], a2y = d[4], a2z = d[5];
    float n1 = rsqrtf(a1x * a1x + a1y * a1y + a1z * a1z);
    float b1x = a1x * n1, b1y = a1y * n1, b1z = a1z * n1;
    float dp = b1x * a2x + b1y * a2y + b1z * a2z;
    float c2x = a2x - dp * b1x, c2y = a2y - dp * b1y, c2z = a2z - dp * b1z;
    float n2 = rsqrtf(c2x * c2x + c2y * c2y + c2z * c2z);
    float b2x = c2x * n2, b2y = c2y * n2, b2z = c2z * n2;
    float b3x = b1y * b2z - b1z * b2y;
    float b3y = b1z * b2x - b1x * b2z;
    float b3z = b1x * b2y - b1y * b2x;
    float* o = rot + (size_t)i * 9;
    o[0] = b1x; o[1] = b1y; o[2] = b1z;
    o[3] = b2x; o[4] = b2y; o[5] = b2z;
    o[6] = b3x; o[7] = b3y; o[8] = b3z;
}

// ---------------------------------------------------------------------------
// 7. vel_world = root[:, :-1] @ pred_vel ; root_world = R_flip @ root[:, 1:]
// ---------------------------------------------------------------------------
__global__ void k_vel_root(const float* __restrict__ rot, const float* __restrict__ pvel,
                           const int* __restrict__ yup,
                           float* __restrict__ vel_ws, float* __restrict__ root_out) {
    int i = blockIdx.x * blockDim.x + threadIdx.x;
    if (i >= TOK) return;
    int b = i / F_, f = i % F_;
    const float* Rm = rot + ((size_t)b * (F_ + 1) + f) * 9;
    const float* pv = pvel + (size_t)i * 3;
#pragma unroll
    for (int r = 0; r < 3; ++r)
        vel_ws[(size_t)i * 3 + r] = Rm[r * 3] * pv[0] + Rm[r * 3 + 1] * pv[1] + Rm[r * 3 + 2] * pv[2];
    const float* Rn = rot + ((size_t)b * (F_ + 1) + f + 1) * 9;
    int flip = *yup;
#pragma unroll
    for (int r = 0; r < 3; ++r)
#pragma unroll
        for (int c = 0; c < 3; ++c)
            root_out[(size_t)i * 9 + r * 3 + c] = (flip && r > 0) ? -Rn[r * 3 + c] : Rn[r * 3 + c];
}

// ---------------------------------------------------------------------------
// 8. per-batch cumulative sum of vel_world + y-up flip
// ---------------------------------------------------------------------------
__global__ void k_cumsum(const float* __restrict__ vel_ws, const int* __restrict__ yup,
                         float* __restrict__ trans_out) {
    int b = blockIdx.x;
    int comp = threadIdx.x;
    if (comp >= 3) return;
    int flip = *yup;
    float s = 0.0f;
    for (int f = 0; f < F_; ++f) {
        s += vel_ws[((size_t)b * F_ + f) * 3 + comp];
        trans_out[((size_t)b * F_ + f) * 3 + comp] = (flip && comp > 0) ? -s : s;
    }
}

// ---------------------------------------------------------------------------
// launcher
// ---------------------------------------------------------------------------
extern "C" void kernel_launch(void* const* d_in, const int* in_sizes, int n_in,
                              void* d_out, int out_size, void* d_ws, size_t ws_size,
                              hipStream_t stream) {
    (void)in_sizes; (void)n_in; (void)out_size; (void)ws_size;
    const float* context  = (const float*)d_in[0];
    const float* pred_vel = (const float*)d_in[1];
    const float* r6d      = (const float*)d_in[3];
    const float* feet     = (const float*)d_in[4];
    const float* contact  = (const float*)d_in[5];
    const int*   yup      = (const int*)d_in[7];
    const float* W_ih0 = (const float*)d_in[8];
    const float* W_hh0 = (const float*)d_in[9];
    const float* b_ih0 = (const float*)d_in[10];
    const float* b_hh0 = (const float*)d_in[11];
    const float* W_ih1 = (const float*)d_in[12];
    const float* W_hh1 = (const float*)d_in[13];
    const float* b_ih1 = (const float*)d_in[14];
    const float* b_hh1 = (const float*)d_in[15];
    const float* dec0_w = (const float*)d_in[16];
    const float* dec0_b = (const float*)d_in[17];
    const float* dec1_w = (const float*)d_in[18];
    const float* dec1_b = (const float*)d_in[19];

    char* ws = (char*)d_ws;
    size_t off = 0;
    auto take = [&](size_t bytes) { size_t o = off; off = (off + bytes + 255) & ~(size_t)255; return o; };
    size_t o_bar  = take(16 * sizeof(unsigned int));
    size_t o_xc   = take((size_t)TOK * KPAD0 * 2);
    size_t o_wih0 = take((size_t)NG_ * KPAD0 * 2);
    size_t o_whh0 = take((size_t)NG_ * H_ * 2);
    size_t o_wih1 = take((size_t)NG_ * H_ * 2);
    size_t o_whh1 = take((size_t)NG_ * H_ * 2);
    size_t o_xw   = take((size_t)TOK * NG_ * 4);
    size_t o_h1   = take((size_t)TOK * H_ * 2);
    size_t o_h2   = take((size_t)TOK * H_ * 2);
    size_t o_hsh  = take((size_t)2 * 16 * H_ * 2);
    size_t o_rot  = take((size_t)B_ * (F_ + 1) * 9 * 4);
    size_t o_vel  = take((size_t)TOK * 3 * 4);

    unsigned int* bars = (unsigned int*)(ws + o_bar);
    __bf16* xc   = (__bf16*)(ws + o_xc);
    __bf16* wih0 = (__bf16*)(ws + o_wih0);
    __bf16* whh0 = (__bf16*)(ws + o_whh0);
    __bf16* wih1 = (__bf16*)(ws + o_wih1);
    __bf16* whh1 = (__bf16*)(ws + o_whh1);
    float*  xw   = (float*)(ws + o_xw);
    __bf16* h1   = (__bf16*)(ws + o_h1);
    __bf16* h2   = (__bf16*)(ws + o_h2);
    __bf16* hsh  = (__bf16*)(ws + o_hsh);
    float*  rot  = (float*)(ws + o_rot);
    float*  velb = (float*)(ws + o_vel);

    float* out_pred_root = (float*)d_out;                         // 32*2049*6
    float* out_pred_vel  = out_pred_root + (size_t)B_ * (F_ + 1) * 6;
    float* out_root      = out_pred_vel + (size_t)TOK * 3;        // 32*2048*9
    float* out_trans     = out_root + (size_t)TOK * 9;            // 32*2048*3

    // 0. zero barriers
    k_zero_u32<<<1, 32, 0, stream>>>(bars, 16);

    // 1. weight conversions
    {
        size_t n0 = (size_t)NG_ * KPAD0;
        k_f32_to_bf16_pad<<<(unsigned)((n0 + 255) / 256), 256, 0, stream>>>(W_ih0, wih0, NG_, DIN_, KPAD0);
        size_t n1 = (size_t)NG_ * H_;
        unsigned g1 = (unsigned)((n1 + 255) / 256);
        k_f32_to_bf16_pad<<<g1, 256, 0, stream>>>(W_hh0, whh0, NG_, H_, H_);
        k_f32_to_bf16_pad<<<g1, 256, 0, stream>>>(W_ih1, wih1, NG_, H_, H_);
        k_f32_to_bf16_pad<<<g1, 256, 0, stream>>>(W_hh1, whh1, NG_, H_, H_);
    }

    // 2. feature assembly
    k_assemble_xc<<<TOK, 256, 0, stream>>>(context, pred_vel, r6d, feet, contact, xc);

    // 3. layer-0 input GEMM: xw = xc @ W_ih0^T + b_ih0 + b_hh0
    k_gemm_bf16_nt<<<dim3(NG_ / 128, TOK / 128), 256, 0, stream>>>(
        xc, wih0, b_ih0, b_hh0, xw, TOK, NG_, KPAD0);

    // 4. layer-0 recurrence
    k_lstm_scan<<<32, 256, SCAN_SMEM, stream>>>(xw, whh0, hsh, h1, bars + 0);

    // 5. layer-1 input GEMM: xw = h1 @ W_ih1^T + b_ih1 + b_hh1
    k_gemm_bf16_nt<<<dim3(NG_ / 128, TOK / 128), 256, 0, stream>>>(
        h1, wih1, b_ih1, b_hh1, xw, TOK, NG_, H_);

    // 6. layer-1 recurrence
    k_lstm_scan<<<32, 256, SCAN_SMEM, stream>>>(xw, whh1, hsh, h2, bars + 2);

    // 7. decoders -> pred_root / pred_vel outputs
    k_decode<<<TOK / 8, 256, 0, stream>>>(h2, r6d, pred_vel, dec0_w, dec0_b, dec1_w, dec1_b,
                                          out_pred_root, out_pred_vel);

    // 8. rotations
    k_r6d<<<(B_ * (F_ + 1) + 255) / 256, 256, 0, stream>>>(out_pred_root, rot);

    // 9. world velocity + root_world
    k_vel_root<<<TOK / 256, 256, 0, stream>>>(rot, out_pred_vel, yup, velb, out_root);

    // 10. cumulative translation
    k_cumsum<<<B_, 4, 0, stream>>>(velb, yup, out_trans);
}
